// TrajectoryPredictionModel_41970420417898
// MI455X (gfx1250) — matrix operand order repaired and checked by reading, not compiled, not run on hardware
//
#include <hip/hip_runtime.h>

// ---------------------------------------------------------------------------
// Fused trajectory-prediction pipeline for MI455X (gfx1250, wave32, WMMA).
// One block == one scene (64 pedestrian rows). Entire pipeline (8-step LSTM
// encoder -> dense "GCN" + BN -> 1-step temp LSTM -> 1-step decoder LSTM ->
// transposed output) runs out of LDS; only x is read and out is written.
// All matmuls use v_wmma_f32_16x16x32_f16 with f32 accumulation.
// ---------------------------------------------------------------------------

typedef _Float16 v16h __attribute__((ext_vector_type(16)));
typedef _Float16 v8h  __attribute__((ext_vector_type(8)));
typedef _Float16 v2h  __attribute__((ext_vector_type(2)));
typedef float    v8f  __attribute__((ext_vector_type(8)));

#define NSCENES 1024
#define KDIM    128      // all four matmuls have K = 128

// Dynamic-LDS layout (bytes)
#define OFF_WT    0        // max 512*128 f16  = 131072 (shared weight region)
#define OFF_HBUF  131072   // 64*128 f16       =  16384
#define OFF_GBUF  147456   // 64*128 f16       =  16384
#define OFF_ZDEC  163840   // 64*96  f32       =  24576
#define OFF_XS    188416   // 8*64*2 f32       =   4096
#define OFF_WENC  192512   // 2*512  f32       =   4096
#define OFF_BENC  196608   // 512    f32       =   2048
#define OFF_BTMP  198656   // 512    f32       =   2048
#define OFF_BNA   200704   // 128    f32       =    512
#define OFF_BNB   201216   // 128    f32       =    512
#define OFF_BDEC  201728   // 96     f32       =    384
#define SMEM_BYTES 202112

union AF { v16h v; v8h h[2]; };

__device__ __forceinline__ float fsig(float x) {
  return __builtin_amdgcn_rcpf(1.0f + __expf(-x));
}
__device__ __forceinline__ float ftanhf(float x) {
  return 1.0f - 2.0f * __builtin_amdgcn_rcpf(1.0f + __expf(2.0f * x));
}

// A fragment: 16x32 f16 tile of row-major [rows][128] f16 buffer.
// lane L: M = mBase + L%16 ; elems 0..7 -> K = kBase + half*8 + 0..7,
// elems 8..15 -> K = kBase + 16 + half*8 + 0..7. Both 16B-aligned runs.
__device__ __forceinline__ v16h load_a(const _Float16* __restrict__ buf,
                                       int mBase, int kBase, int lane16, int hf) {
  AF a;
  const _Float16* p0 = buf + (mBase + lane16) * KDIM + kBase + hf * 8;
  a.h[0] = *(const v8h*)p0;
  a.h[1] = *(const v8h*)(p0 + 16);
  return a.v;
}

// B fragment: 32x16 f16 from N-major WT[Ncol][128]:
// lane L: N = nBase + L%16 ; elems 0..7 -> K = kBase + half*8 + 0..7,
// elems 8..15 -> K = kBase + 16 + half*8 + 0..7 (matches load_a).
__device__ __forceinline__ v16h load_b(const _Float16* __restrict__ wT,
                                       int nBase, int kBase, int lane16, int hf) {
  AF b;
  const _Float16* p0 = wT + (nBase + lane16) * KDIM + kBase + hf * 8;
  b.h[0] = *(const v8h*)p0;
  b.h[1] = *(const v8h*)(p0 + 16);
  return b.v;
}

__device__ __forceinline__ v8f wmma_f(v16h a, v16h b, v8f c) {
  return __builtin_amdgcn_wmma_f32_16x16x32_f16(false, a, false, b,
                                                (short)0, c, false, false);
}

// Stage W[k][n] (f32, k<128) into LDS transposed+converted: dst[n*128+k] (f16).
// Coalesced global reads (consecutive n), packed 32-bit LDS writes (k pairs).
__device__ void load_wT(const float* __restrict__ W, int Ncol,
                        _Float16* __restrict__ dst, int tid) {
  int total = 64 * Ncol;  // (k-pair, n) items
#pragma unroll 1
  for (int i = tid; i < total; i += 256) {
    int n  = i % Ncol;
    int k2 = (i / Ncol) * 2;
    float a = W[k2 * Ncol + n];
    float b = W[(k2 + 1) * Ncol + n];
    v2h pk;
    pk.x = (_Float16)a;
    pk.y = (_Float16)b;
    *(v2h*)(dst + n * KDIM + k2) = pk;
  }
}

__global__ void __launch_bounds__(256) traj_fused_kernel(
    const float* __restrict__ x,      const float* __restrict__ W_enc,
    const float* __restrict__ U_enc,  const float* __restrict__ b_enc,
    const float* __restrict__ W_gcn,  const float* __restrict__ b_gcn,
    const float* __restrict__ bn_gamma, const float* __restrict__ bn_beta,
    const float* __restrict__ bn_mean,  const float* __restrict__ bn_var,
    const float* __restrict__ W_temp, const float* __restrict__ b_temp,
    const float* __restrict__ W_dec,  const float* __restrict__ b_dec,
    float* __restrict__ out) {
  extern __shared__ char smem[];
  _Float16* wT   = (_Float16*)(smem + OFF_WT);
  _Float16* hbuf = (_Float16*)(smem + OFF_HBUF);
  _Float16* gbuf = (_Float16*)(smem + OFF_GBUF);
  float*    zdec = (float*)(smem + OFF_ZDEC);
  float*    xs   = (float*)(smem + OFF_XS);
  float*    wenc = (float*)(smem + OFF_WENC);
  float*    benc = (float*)(smem + OFF_BENC);
  float*    btmp = (float*)(smem + OFF_BTMP);
  float*    bnA  = (float*)(smem + OFF_BNA);
  float*    bnB  = (float*)(smem + OFF_BNB);
  float*    bdec = (float*)(smem + OFF_BDEC);

  const int tid    = threadIdx.x;
  const int lane   = tid & 31;
  const int w      = tid >> 5;       // wave id 0..7
  const int lane16 = lane & 15;
  const int hf     = lane >> 4;      // half-wave select
  const int mBase  = (w & 3) * 16;   // row subtile for this wave
  const int grp    = w >> 2;         // column group (0 or 1)
  const int scene  = blockIdx.x;

  // Hint the weight working set into cache (global_prefetch_b8).
  __builtin_prefetch(U_enc + tid, 0, 3);
  __builtin_prefetch(W_temp + tid, 0, 3);
  __builtin_prefetch(W_gcn + tid, 0, 2);
  __builtin_prefetch(W_dec + tid, 0, 2);

  // ---------------- phase 0: stage parameters / x tile / clear h ----------
#pragma unroll 1
  for (int i = tid; i < 1024; i += 256) xs[i]   = x[scene * 1024 + i];
#pragma unroll 1
  for (int i = tid; i < 1024; i += 256) wenc[i] = W_enc[i];
#pragma unroll 1
  for (int i = tid; i < 512; i += 256)  benc[i] = b_enc[i];
#pragma unroll 1
  for (int i = tid; i < 512; i += 256)  btmp[i] = b_temp[i];
  if (tid < 96) bdec[tid] = b_dec[tid];
  if (tid < 128) {
    float sc = bn_gamma[tid] * rsqrtf(bn_var[tid] + 1e-3f);
    bnA[tid] = sc;
    bnB[tid] = (b_gcn[tid] - bn_mean[tid]) * sc + bn_beta[tid];
  }
#pragma unroll 1
  for (int i = tid; i < 4096; i += 256) ((unsigned int*)hbuf)[i] = 0u;
  load_wT(U_enc, 512, wT, tid);  // U_enc^T as f16 [512][128]
  __syncthreads();

  // ---------------- encoder LSTM: 8 recurrent steps -----------------------
  v8f cst[4];  // cell state, cols (grp*4+ic)*16 + lane16, rows mBase+r+8*hf
#pragma unroll
  for (int ic = 0; ic < 4; ++ic)
#pragma unroll
    for (int r = 0; r < 8; ++r) cst[ic][r] = 0.0f;

#pragma unroll 1
  for (int t = 0; t < 8; ++t) {
    // rank-2 input projection folded into accumulator init
    float xa[8], xb[8];
#pragma unroll
    for (int r = 0; r < 8; ++r) {
      int p = mBase + r + 8 * hf;
      xa[r] = xs[t * 128 + p * 2 + 0];
      xb[r] = xs[t * 128 + p * 2 + 1];
    }
    v8f acc[4][4];
#pragma unroll
    for (int ic = 0; ic < 4; ++ic) {
#pragma unroll
      for (int gt = 0; gt < 4; ++gt) {
        int n   = gt * 128 + (grp * 4 + ic) * 16 + lane16;
        float bb = benc[n], w0 = wenc[n], w1 = wenc[512 + n];
#pragma unroll
        for (int r = 0; r < 8; ++r) acc[ic][gt][r] = bb + xa[r] * w0 + xb[r] * w1;
      }
    }
    // A fragments of h (zero for t==0 via cleared hbuf)
    v16h af[4];
#pragma unroll
    for (int ks = 0; ks < 4; ++ks) af[ks] = load_a(hbuf, mBase, ks * 32, lane16, hf);
    __syncthreads();  // all waves done reading h before it is rewritten

    // z += h @ U_enc : 4 ic x 4 gates x 4 k-steps = 64 WMMAs / wave / step
#pragma unroll
    for (int ic = 0; ic < 4; ++ic) {
      int icg = grp * 4 + ic;
#pragma unroll
      for (int gt = 0; gt < 4; ++gt) {
        int nb = (gt * 8 + icg) * 16;
#pragma unroll
        for (int ks = 0; ks < 4; ++ks)
          acc[ic][gt] = wmma_f(af[ks], load_b(wT, nb, ks * 32, lane16, hf), acc[ic][gt]);
      }
    }
    // fused keras-order gates: i,f,g,o ; c = f*c + i*g ; h = o*tanh(c)
#pragma unroll
    for (int ic = 0; ic < 4; ++ic) {
#pragma unroll
      for (int r = 0; r < 8; ++r) {
        float iv = fsig(acc[ic][0][r]);
        float fv = fsig(acc[ic][1][r]);
        float gv = ftanhf(acc[ic][2][r]);
        float ov = fsig(acc[ic][3][r]);
        float c  = fv * cst[ic][r] + iv * gv;
        cst[ic][r]    = c;
        acc[ic][0][r] = ov * ftanhf(c);
      }
    }
#pragma unroll
    for (int ic = 0; ic < 4; ++ic) {
      int col = (grp * 4 + ic) * 16 + lane16;
#pragma unroll
      for (int r = 0; r < 8; ++r)
        hbuf[(mBase + r + 8 * hf) * KDIM + col] = (_Float16)acc[ic][0][r];
    }
    __syncthreads();
  }

  // ---------------- GCN projection + folded BatchNorm ---------------------
  load_wT(W_gcn, 128, wT, tid);
  __syncthreads();
  {
    v8f acc[4];
#pragma unroll
    for (int q = 0; q < 4; ++q)
#pragma unroll
      for (int r = 0; r < 8; ++r) acc[q][r] = 0.0f;
    v16h af[4];
#pragma unroll
    for (int ks = 0; ks < 4; ++ks) af[ks] = load_a(hbuf, mBase, ks * 32, lane16, hf);
#pragma unroll
    for (int q = 0; q < 4; ++q) {
      int nb = (grp * 4 + q) * 16;
#pragma unroll
      for (int ks = 0; ks < 4; ++ks)
        acc[q] = wmma_f(af[ks], load_b(wT, nb, ks * 32, lane16, hf), acc[q]);
    }
#pragma unroll
    for (int q = 0; q < 4; ++q) {
      int n = (grp * 4 + q) * 16 + lane16;
      float sA = bnA[n], sB = bnB[n];
#pragma unroll
      for (int r = 0; r < 8; ++r)
        gbuf[(mBase + r + 8 * hf) * KDIM + n] = (_Float16)(acc[q][r] * sA + sB);
    }
  }
  __syncthreads();  // wT reads + gbuf writes complete

  // ---------------- temp LSTM, single step (h0 = c0 = 0) ------------------
  load_wT(W_temp, 512, wT, tid);
  __syncthreads();
  {
    v8f acc[4][4];
#pragma unroll
    for (int ic = 0; ic < 4; ++ic)
#pragma unroll
      for (int gt = 0; gt < 4; ++gt) {
        int n = gt * 128 + (grp * 4 + ic) * 16 + lane16;
        float bb = btmp[n];
#pragma unroll
        for (int r = 0; r < 8; ++r) acc[ic][gt][r] = bb;
      }
    v16h af[4];
#pragma unroll
    for (int ks = 0; ks < 4; ++ks) af[ks] = load_a(gbuf, mBase, ks * 32, lane16, hf);
#pragma unroll
    for (int ic = 0; ic < 4; ++ic) {
      int icg = grp * 4 + ic;
#pragma unroll
      for (int gt = 0; gt < 4; ++gt) {
        int nb = (gt * 8 + icg) * 16;
#pragma unroll
        for (int ks = 0; ks < 4; ++ks)
          acc[ic][gt] = wmma_f(af[ks], load_b(wT, nb, ks * 32, lane16, hf), acc[ic][gt]);
      }
    }
#pragma unroll
    for (int ic = 0; ic < 4; ++ic) {
      int col = (grp * 4 + ic) * 16 + lane16;
#pragma unroll
      for (int r = 0; r < 8; ++r) {
        float iv = fsig(acc[ic][0][r]);
        float gv = ftanhf(acc[ic][2][r]);
        float ov = fsig(acc[ic][3][r]);
        float c  = iv * gv;  // f*c0 vanishes
        hbuf[(mBase + r + 8 * hf) * KDIM + col] = (_Float16)(ov * ftanhf(c));
      }
    }
  }
  __syncthreads();

  // ---------------- decoder LSTM single step + transposed output ----------
  load_wT(W_dec, 96, wT, tid);
  __syncthreads();
  {
#pragma unroll
    for (int q = 0; q < 3; ++q) {          // 24 tiles (4m x 6n), 3 per wave
      int id = w * 3 + q;
      int ms = (id / 6) * 16;
      int nb = (id % 6) * 16;
      int n  = nb + lane16;
      float bb = bdec[n];
      v8f acc;
#pragma unroll
      for (int r = 0; r < 8; ++r) acc[r] = bb;
#pragma unroll
      for (int ks = 0; ks < 4; ++ks)
        acc = wmma_f(load_a(hbuf, ms, ks * 32, lane16, hf),
                     load_b(wT, nb, ks * 32, lane16, hf), acc);
#pragma unroll
      for (int r = 0; r < 8; ++r)
        zdec[(ms + r + 8 * hf) * 96 + n] = acc[r];
    }
  }
  __syncthreads();
#pragma unroll 1
  for (int e = tid; e < 64 * 24; e += 256) {
    int p = e / 24, j = e % 24;
    float iv = fsig(zdec[p * 96 + j]);
    float gv = ftanhf(zdec[p * 96 + 48 + j]);
    float ov = fsig(zdec[p * 96 + 72 + j]);
    float hd = ov * ftanhf(iv * gv);
    // out[n, j/2, p, j%2] with shape [1024, 12, 64, 2]
    out[((scene * 12 + (j >> 1)) * 64 + p) * 2 + (j & 1)] = hd;
  }
}

extern "C" void kernel_launch(void* const* d_in, const int* in_sizes, int n_in,
                              void* d_out, int out_size, void* d_ws, size_t ws_size,
                              hipStream_t stream) {
  const float* x        = (const float*)d_in[0];
  const float* W_enc    = (const float*)d_in[1];
  const float* U_enc    = (const float*)d_in[2];
  const float* b_enc    = (const float*)d_in[3];
  const float* W_gcn    = (const float*)d_in[4];
  const float* b_gcn    = (const float*)d_in[5];
  const float* bn_gamma = (const float*)d_in[6];
  const float* bn_beta  = (const float*)d_in[7];
  const float* bn_mean  = (const float*)d_in[8];
  const float* bn_var   = (const float*)d_in[9];
  const float* W_temp   = (const float*)d_in[10];
  const float* b_temp   = (const float*)d_in[11];
  const float* W_dec    = (const float*)d_in[12];
  const float* b_dec    = (const float*)d_in[13];
  float* out = (float*)d_out;

  (void)in_sizes; (void)n_in; (void)out_size; (void)d_ws; (void)ws_size;

  // Allow large dynamic LDS (MI455X WGP: 320 KB); best-effort, capture-safe.
  (void)hipFuncSetAttribute((const void*)traj_fused_kernel,
                            hipFuncAttributeMaxDynamicSharedMemorySize,
                            (int)SMEM_BYTES);

  traj_fused_kernel<<<NSCENES, 256, SMEM_BYTES, stream>>>(
      x, W_enc, U_enc, b_enc, W_gcn, b_gcn, bn_gamma, bn_beta, bn_mean, bn_var,
      W_temp, b_temp, W_dec, b_dec, out);
}
